// PatchAttnClassifierNeck_28595892257368
// MI455X (gfx1250) — compile-verified, matrix operand order checked by reading, not run on hardware
//
#include <hip/hip_runtime.h>
#include <hip/hip_bf16.h>

// ---------------- problem constants ----------------
constexpr int NB   = 2;     // batch
constexpr int C    = 512;   // channels
constexpr int DD   = 32;    // depth
constexpr int HH   = 16, WW = 16;
constexpr int HP   = 14;    // pooled H/W
constexpr int HWR  = 196;   // real pooled spatial size
constexpr int NP   = 224;   // padded spatial (14 tiles of 16)
constexpr int NO   = 4;     // num_outs
constexpr int KD   = 8;     // key_dim
constexpr int CO   = NO * C;        // 2048
constexpr int NDT  = NB * DD;       // 64 (n,d) pairs

typedef __bf16 bf16;
typedef __attribute__((ext_vector_type(8)))  __bf16 v8bf;
typedef __attribute__((ext_vector_type(16))) __bf16 v16bf;
typedef __attribute__((ext_vector_type(8)))  float  v8f;
typedef __attribute__((ext_vector_type(4)))  unsigned int u32x4;
typedef __attribute__((ext_vector_type(8)))  int i32x8;
typedef __attribute__((ext_vector_type(4)))  int i32x4;

// ---------------- workspace layout (bytes) ----------------
constexpr size_t SZ_XPT = (size_t)NDT * NP * C * 2;     // pooled acts, bf16, [nd][hw][c]
constexpr size_t SZ_WVB = (size_t)CO * C * 2;           // Wv bf16
constexpr size_t SZ_WKB = (size_t)(NO*KD) * C * 2;      // Wk bf16
constexpr size_t SZ_XM  = (size_t)NB * C * DD * 4;      // spatial mean f32
constexpr size_t SZ_Q   = (size_t)NB * (NO*KD) * DD * 4;// query f32
constexpr size_t SZ_AP  = (size_t)NDT * NO * NP * 4;    // padded attn f32
constexpr size_t OFF_XPT = 0;
constexpr size_t OFF_WVB = OFF_XPT + SZ_XPT;
constexpr size_t OFF_WKB = OFF_WVB + SZ_WVB;
constexpr size_t OFF_XM  = OFF_WKB + SZ_WKB;
constexpr size_t OFF_Q   = OFF_XM  + SZ_XM;
constexpr size_t OFF_AP  = OFF_Q   + SZ_Q;
constexpr size_t OFF_OM  = OFF_AP  + SZ_AP;             // out_mid (N,4,512,D) f32

// ------------- WMMA fragment load (bf16, 16x32 tile, row-major, ld = K stride) -------------
// A 16x32 layout (ISA 7.12.2): lanes 0-15 hold rows M=0..15 with K = {0..7,16..23};
// lanes 16-31 hold K = {8..15,24..31}.  B (K x 16, stored transposed as [col][k]) mirrors it.
__device__ __forceinline__ v16bf load_frag_bf16(const bf16* base, int ld, int row, int k0, int lane) {
    int sel = (lane >> 4) & 1;
    const bf16* p = base + (size_t)row * ld + k0 + sel * 8;
    v8bf lo = *(const v8bf*)(p);
    v8bf hi = *(const v8bf*)(p + 16);
    return __builtin_shufflevector(lo, hi, 0,1,2,3,4,5,6,7,8,9,10,11,12,13,14,15);
}

// LDS-staged fragment: slice layout [col(0..223)][k(0..31)]
__device__ __forceinline__ v16bf load_frag_lds(const bf16* bs, int col, int lane) {
    int sel = (lane >> 4) & 1;
    const bf16* p = bs + col * 32 + sel * 8;
    v8bf lo = *(const v8bf*)(p);
    v8bf hi = *(const v8bf*)(p + 16);
    return __builtin_shufflevector(lo, hi, 0,1,2,3,4,5,6,7,8,9,10,11,12,13,14,15);
}

// ------------- Tensor Data Mover: async DMA of a 32(K) x 224(N) bf16 slice into LDS -------------
#if defined(__has_builtin)
#if __has_builtin(__builtin_amdgcn_tensor_load_to_lds) && __has_builtin(__builtin_amdgcn_s_wait_tensorcnt)
#define HAVE_TDM 1
#endif
#endif

#ifdef HAVE_TDM
__device__ __forceinline__ void tdm_load_b_slice(const bf16* gsrc, unsigned lds_off) {
    unsigned long long ga = (unsigned long long)(uintptr_t)gsrc;
    // D# group 0: count=1, lds_addr, global_addr[56:0], type=2 ("image")
    u32x4 g0 = {};
    g0[0] = 1u;                                            // count = 1 (valid descriptor)
    g0[1] = lds_off;                                       // lds_addr (bytes)
    g0[2] = (unsigned)(ga & 0xFFFFFFFFu);                  // global_addr[31:0]
    g0[3] = (unsigned)((ga >> 32) & 0x01FFFFFFu)           // global_addr[56:32]
          | (2u << 30);                                    // type = 2
    // D# group 1: data_size=1 (2B), tensor_dim0=512, tensor_dim1=224,
    //             tile_dim0=32, tile_dim1=224, tensor_dim0_stride=512
    i32x8 g1 = {};
    g1[0] = (int)(1u << 16);                               // data_size = 2 bytes, wg_mask = 0
    g1[1] = (int)(512u << 16);                             // tensor_dim0[15:0] at bits[63:48]
    g1[2] = (int)(224u << 16);                             // tensor_dim1[15:0] at bits[95:80]
    g1[3] = (int)(32u  << 16);                             // tile_dim0 at bits[127:112]
    g1[4] = 224;                                           // tile_dim1 at bits[143:128]
    g1[5] = 512;                                           // tensor_dim0_stride[31:0]
    i32x4 z4 = {};
#if __clang_major__ >= 23
    i32x8 z8 = {};
    __builtin_amdgcn_tensor_load_to_lds(g0, g1, z4, z4, z8, 0);
#else
    __builtin_amdgcn_tensor_load_to_lds(g0, g1, z4, z4, 0);
#endif
}
#endif

// ---------------- K0: weight conversion to bf16 ----------------
__global__ void k_convert(const float* __restrict__ Wv, const float* __restrict__ Wk,
                          bf16* __restrict__ WvB, bf16* __restrict__ WkB) {
    int i = blockIdx.x * 256 + threadIdx.x;
    if (i < CO * C) {
        WvB[i] = (bf16)Wv[i];
    } else {
        int j = i - CO * C;
        if (j < (NO*KD) * C) WkB[j] = (bf16)Wk[j];
    }
}

// ---------------- K1: 3x3 avg-pool + spatial mean ----------------
__global__ __launch_bounds__(256) void k_pool(const float* __restrict__ x,
                                              bf16* __restrict__ xpT,
                                              float* __restrict__ xm) {
    __shared__ float sx[256];
    __shared__ float red[256];
    int blk = blockIdx.x;
    int n = blk / (C * DD);
    int c = (blk / DD) % C;
    int d = blk % DD;
    int tid = threadIdx.x;
    const float* xb = x + (((size_t)n * C + c) * DD + d) * (HH * WW);
    float v = xb[tid];
    sx[tid]  = v;
    red[tid] = v;
    __syncthreads();
    for (int s = 128; s > 0; s >>= 1) {
        if (tid < s) red[tid] += red[tid + s];
        __syncthreads();
    }
    if (tid == 0) xm[((size_t)n * C + c) * DD + d] = red[0] * (1.0f / 256.0f);
    int nd = n * DD + d;
    if (tid < HWR) {
        int i = tid / HP, j = tid % HP;
        float s = 0.f;
        #pragma unroll
        for (int di = 0; di < 3; ++di)
            #pragma unroll
            for (int dj = 0; dj < 3; ++dj)
                s += sx[(i + di) * WW + (j + dj)];
        xpT[((size_t)nd * NP + tid) * C + c] = (bf16)(s * (1.0f / 9.0f));
    } else if (tid < NP) {
        xpT[((size_t)nd * NP + tid) * C + c] = (bf16)0.0f;
    }
}

// ---------------- K2: query = Wq @ xm ----------------
__global__ void k_query(const float* __restrict__ xm, const float* __restrict__ Wq,
                        float* __restrict__ query) {
    int g = blockIdx.x * 256 + threadIdx.x;
    if (g >= NB * (NO*KD) * DD) return;
    int n  = g / ((NO*KD) * DD);
    int kk = (g / DD) % (NO*KD);
    int d  = g % DD;
    float acc = 0.f;
    for (int c = 0; c < C; ++c)
        acc += xm[((size_t)n * C + c) * DD + d] * Wq[kk * C + c];
    query[((size_t)n * (NO*KD) + kk) * DD + d] = acc;
}

// ---------------- K3: k-GEMM (WMMA) + q.k dot + softmax -> attn ----------------
__global__ __launch_bounds__(128) void k_kq(const bf16* __restrict__ WkB,
                                            const bf16* __restrict__ xpT,
                                            const float* __restrict__ query,
                                            float* __restrict__ attnp,
                                            float* __restrict__ attn_out) {
    __shared__ float kxk[NO*KD][NP];
    __shared__ float dotb[NO][NP];
    __shared__ float qv[NO*KD];
    int nd   = blockIdx.x;
    int n    = nd / DD, d = nd % DD;
    int tid  = threadIdx.x;
    int lane = tid & 31;
    int wave = tid >> 5;
    if (tid < NO*KD) qv[tid] = query[((size_t)n * (NO*KD) + tid) * DD + d];
    const bf16* B = xpT + (size_t)nd * NP * C;

    // 2 M-tiles x 14 N-tiles = 28 tiles, 7 per wave (uniform per wave)
    #pragma unroll
    for (int it = 0; it < 7; ++it) {
        int t  = wave + 4 * it;
        int mt = t & 1, nt = t >> 1;
        v8f acc = {};
        for (int k0 = 0; k0 < C; k0 += 32) {
            v16bf a = load_frag_bf16(WkB, C, mt * 16 + (lane & 15), k0, lane);
            v16bf b = load_frag_bf16(B,   C, nt * 16 + (lane & 15), k0, lane);
            acc = __builtin_amdgcn_wmma_f32_16x16x32_bf16(false, a, false, b,
                                                          (short)0, acc, false, false);
        }
        int cr = (lane >> 4) * 8, cc = lane & 15;
        #pragma unroll
        for (int r = 0; r < 8; ++r)
            kxk[mt * 16 + cr + r][nt * 16 + cc] = acc[r];
    }
    __syncthreads();

    const float temperature = 0.35355339059327373f; // sqrt(1/8)
    for (int idx = tid; idx < NO * NP; idx += 128) {
        int o = idx / NP, hw = idx % NP;
        float s = 0.f;
        #pragma unroll
        for (int j = 0; j < KD; ++j) s += qv[o * KD + j] * kxk[o * KD + j][hw];
        dotb[o][hw] = s * temperature;
    }
    __syncthreads();

    // softmax over 196 real columns: wave o owns row o
    {
        int o = wave;
        float m = -3.4e38f;
        for (int hw = lane; hw < HWR; hw += 32) m = fmaxf(m, dotb[o][hw]);
        #pragma unroll
        for (int off = 16; off > 0; off >>= 1) m = fmaxf(m, __shfl_xor(m, off, 32));
        float s = 0.f;
        for (int hw = lane; hw < HWR; hw += 32) {
            float e = __expf(dotb[o][hw] - m);
            dotb[o][hw] = e;
            s += e;
        }
        #pragma unroll
        for (int off = 16; off > 0; off >>= 1) s += __shfl_xor(s, off, 32);
        float inv = 1.0f / s;
        for (int hw = lane; hw < HWR; hw += 32) {
            float a = dotb[o][hw] * inv;
            attnp[((size_t)nd * NO + o) * NP + hw] = a;
            attn_out[(((size_t)n * NO + o) * DD + d) * HWR + hw] = a;
        }
        for (int hw = HWR + lane; hw < NP; hw += 32)
            attnp[((size_t)nd * NO + o) * NP + hw] = 0.f;
    }
}

// ---------------- K4: big v-GEMM (WMMA + TDM staging) + instance-norm + attn sum + GELU ----
// one block (8 waves) per (nd, 64-row slab of Wv): 64x224 tile, K=512.
// B slices (32K x 224N bf16 = 14 KB) are DMA'd into LDS with tensor_load_to_lds,
// double-buffered and overlapped with the WMMA K-loop.
struct SmemU {
    union {
        bf16  bst[2][32 * NP];   // 28.7 KB staged B slices (live during K loop)
        float vbuf[64][NP];      // 57.3 KB result tile     (live after K loop)
    };
};

__global__ __launch_bounds__(256) void k_v(const bf16* __restrict__ WvB,
                                           const bf16* __restrict__ xpT,
                                           const float* __restrict__ attnp,
                                           const float* __restrict__ gamma,
                                           const float* __restrict__ beta,
                                           float* __restrict__ omid) {
    __shared__ SmemU u;
    __shared__ float arow[NP];
    __shared__ float ps[64][4];
    __shared__ float pq[64][4];
    __shared__ float mrow[64], rrow[64], grow[64], brow[64];

    int blk  = blockIdx.x;
    int nd   = blk >> 5;        // / 32 slabs
    int mblk = blk & 31;
    int n    = nd >> 5, d = nd & 31;
    int o    = mblk >> 3;       // (mblk*64)/512
    int tid  = threadIdx.x;
    int lane = tid & 31;
    int wave = tid >> 5;
    const bf16* A = WvB + (size_t)mblk * 64 * C;
    const bf16* B = xpT + (size_t)nd * NP * C;

    int mt  = wave & 3;   // 4 M-tiles
    int par = wave >> 2;  // N-tile parity: 7 tiles each (constant trip count)
    v8f acc[7];
    #pragma unroll
    for (int i = 0; i < 7; ++i) acc[i] = (v8f){};

    constexpr int KSTEPS = C / 32;               // 16
    constexpr unsigned SLICE_BYTES = 32 * NP * 2; // 14336

#ifdef HAVE_TDM
    unsigned lds_base = (unsigned)(uintptr_t)(&u.bst[0][0]);
    if (wave == 0) tdm_load_b_slice(B, lds_base);            // prefetch slice 0
    for (int kk = 0; kk < KSTEPS; ++kk) {
        if (wave == 0) {
            __builtin_amdgcn_s_wait_tensorcnt(0);            // slice kk arrived
            if (kk + 1 < KSTEPS)                             // kick off next DMA
                tdm_load_b_slice(B + (size_t)(kk + 1) * 32,
                                 lds_base + (unsigned)((kk + 1) & 1) * SLICE_BYTES);
        }
        __syncthreads();                                     // slice kk visible to all waves
        const bf16* bs = u.bst[kk & 1];
        v16bf a = load_frag_bf16(A, C, mt * 16 + (lane & 15), kk * 32, lane);
        #pragma unroll
        for (int i = 0; i < 7; ++i) {
            int nt = par + 2 * i;
            v16bf b = load_frag_lds(bs, nt * 16 + (lane & 15), lane);
            acc[i] = __builtin_amdgcn_wmma_f32_16x16x32_bf16(false, a, false, b,
                                                             (short)0, acc[i], false, false);
        }
        __syncthreads();                                     // slice consumed before overwrite
    }
#else
    // fallback: cooperative global->LDS staging (b128 copies), same double buffering
    for (int kk = 0; kk < KSTEPS; ++kk) {
        bf16* dst = u.bst[kk & 1];
        for (int idx = tid; idx < (32 * NP) / 8; idx += 256) {
            int col = idx / 4, seg = idx & 3;
            *(v8bf*)(dst + col * 32 + seg * 8) =
                *(const v8bf*)(B + (size_t)col * C + kk * 32 + seg * 8);
        }
        __syncthreads();
        const bf16* bs = u.bst[kk & 1];
        v16bf a = load_frag_bf16(A, C, mt * 16 + (lane & 15), kk * 32, lane);
        #pragma unroll
        for (int i = 0; i < 7; ++i) {
            int nt = par + 2 * i;
            v16bf b = load_frag_lds(bs, nt * 16 + (lane & 15), lane);
            acc[i] = __builtin_amdgcn_wmma_f32_16x16x32_bf16(false, a, false, b,
                                                             (short)0, acc[i], false, false);
        }
        __syncthreads();
    }
#endif

    // spill WMMA tiles to LDS result buffer (overlaps staging buffers; loop is done)
    int cr = (lane >> 4) * 8, cc = lane & 15;
    #pragma unroll
    for (int i = 0; i < 7; ++i) {
        int nt = par + 2 * i;
        #pragma unroll
        for (int r = 0; r < 8; ++r)
            u.vbuf[mt * 16 + cr + r][nt * 16 + cc] = acc[i][r];
    }
    if (tid < NP) arow[tid] = attnp[((size_t)nd * NO + o) * NP + tid];
    __syncthreads();

    // instance-norm statistics over the 196 real columns (4 partials per row)
    {
        int row = tid >> 2, part = tid & 3;
        float s = 0.f, q = 0.f;
        for (int cix = part; cix < HWR; cix += 4) {
            float v = u.vbuf[row][cix];
            s += v; q += v * v;
        }
        ps[row][part] = s; pq[row][part] = q;
    }
    __syncthreads();
    if (tid < 64) {
        float s = ps[tid][0] + ps[tid][1] + ps[tid][2] + ps[tid][3];
        float q = pq[tid][0] + pq[tid][1] + pq[tid][2] + pq[tid][3];
        float mu  = s * (1.0f / (float)HWR);
        float var = q * (1.0f / (float)HWR) - mu * mu;
        mrow[tid] = mu;
        rrow[tid] = rsqrtf(var + 1e-5f);
        int c2 = mblk * 64 + tid;
        grow[tid] = gamma[c2];
        brow[tid] = beta[c2];
    }
    __syncthreads();

    // attn-weighted reduction of normalized rows
    {
        int row = tid >> 2, part = tid & 3;
        float mu = mrow[row], rs = rrow[row], g = grow[row], bb = brow[row];
        float w = 0.f;
        for (int cix = part; cix < HWR; cix += 4) {
            float v = (u.vbuf[row][cix] - mu) * rs * g + bb;
            w += v * arow[cix];
        }
        ps[row][part] = w;
    }
    __syncthreads();
    if (tid < 64) {
        float t = ps[tid][0] + ps[tid][1] + ps[tid][2] + ps[tid][3];
        float ge = 0.5f * t * (1.0f + erff(t * 0.70710678118654752f)); // exact GELU
        int c2 = mblk * 64 + tid;
        int c  = c2 & (C - 1);
        omid[(((size_t)n * NO + o) * C + c) * DD + d] = ge;
    }
}

// ---------------- K5: per-output 1x1x1 conv heads ----------------
__global__ void k_heads(const float* __restrict__ omid, const float* __restrict__ Wout,
                        const float* __restrict__ bout, float* __restrict__ outp) {
    int tid = threadIdx.x;                 // 256 = 4*2*32 outputs
    int o = tid >> 6, n = (tid >> 5) & 1, d = tid & 31;
    float acc = bout[o];
    for (int c = 0; c < C; ++c)
        acc += omid[(((size_t)n * NO + o) * C + c) * DD + d] * Wout[o * C + c];
    outp[((size_t)o * NB + n) * DD + d] = acc;
}

// ---------------- launcher ----------------
extern "C" void kernel_launch(void* const* d_in, const int* in_sizes, int n_in,
                              void* d_out, int out_size, void* d_ws, size_t ws_size,
                              hipStream_t stream) {
    const float* x     = (const float*)d_in[0];
    const float* Wq    = (const float*)d_in[1];
    const float* Wk    = (const float*)d_in[2];
    const float* Wv    = (const float*)d_in[3];
    const float* gamma = (const float*)d_in[4];
    const float* beta  = (const float*)d_in[5];
    const float* Wout  = (const float*)d_in[6];
    const float* bout  = (const float*)d_in[7];
    float* out = (float*)d_out;

    char* ws = (char*)d_ws;
    bf16*  xpT   = (bf16*)(ws + OFF_XPT);
    bf16*  WvB   = (bf16*)(ws + OFF_WVB);
    bf16*  WkB   = (bf16*)(ws + OFF_WKB);
    float* xm    = (float*)(ws + OFF_XM);
    float* query = (float*)(ws + OFF_Q);
    float* attnp = (float*)(ws + OFF_AP);
    float* omid  = (float*)(ws + OFF_OM);

    int nconv = CO * C + (NO*KD) * C;
    k_convert<<<(nconv + 255) / 256, 256, 0, stream>>>(Wv, Wk, WvB, WkB);
    k_pool<<<NB * C * DD, 256, 0, stream>>>(x, xpT, xm);
    k_query<<<(NB * (NO*KD) * DD + 255) / 256, 256, 0, stream>>>(xm, Wq, query);
    k_kq<<<NDT, 128, 0, stream>>>(WkB, xpT, query, attnp, out);
    k_v<<<NDT * (CO / 64), 256, 0, stream>>>(WvB, xpT, attnp, gamma, beta, omid);
    k_heads<<<1, 256, 0, stream>>>(omid, Wout, bout, out + (size_t)NB * NO * DD * HWR);
}
